// BiLevelRoutingAttention_42185168781833
// MI455X (gfx1250) — matrix operand order repaired and checked by reading, not compiled, not run on hardware
//
#include <hip/hip_runtime.h>
#include <stdint.h>

// ---------------------------------------------------------------------------
// Problem constants (from reference)
// ---------------------------------------------------------------------------
constexpr int T    = 4;
constexpr int B    = 8;
constexpr int H    = 8;          // heads
constexpr int N    = 1024;       // tokens per (t,b)
constexpr int C    = 256;        // dim
constexpr int D    = 32;         // head dim
constexpr int NW   = 8;          // windows
constexpr int WIN  = 128;        // tokens per window
constexpr int TOPK = 4;
constexpr int TOK  = T * B * N;  // 32768 total tokens
constexpr int C3   = 3 * C;      // 768
constexpr int KEYS = TOPK * WIN; // 512 gathered keys
constexpr float SCALE = 0.17677669529663687f; // 1/sqrt(32)

// ---------------------------------------------------------------------------
// WMMA fragment types (CDNA5 gfx1250, wave32)
// ---------------------------------------------------------------------------
typedef __attribute__((ext_vector_type(16))) __bf16    v16bf;
typedef __attribute__((ext_vector_type(8)))  float     v8f;
typedef __attribute__((ext_vector_type(4)))  uint32_t  u32x4;

union Frag16 {
  v16bf  v;
  __bf16 h[16];
  u32x4  q[2];   // two 16-byte chunks of 8 bf16 each
};

static __device__ inline v8f wmma_bf16(const Frag16& a, const Frag16& b, v8f c) {
  // D = A(16x32 bf16) * B(32x16 bf16) + C(16x16 f32)
  return __builtin_amdgcn_wmma_f32_16x16x32_bf16(
      /*neg_a=*/false, a.v, /*neg_b=*/false, b.v,
      /*c_mod=*/(short)0, c, /*reuse_a=*/false, /*reuse_b=*/false);
}

// A-fragment layout (16-bit A 16x32, as loaded here):
//   lanes 0-15  : M = lane,    K = 0..15
//   lanes 16-31 : M = lane-16, K = 16..31
// B-fragment layout (16-bit B 32x16): lane = column; K run of 16 starting at
//   K = half*16 (VGPR p holds K{2p,2p+1}).
// C/D layout: acc[i] -> M = i + 8*half, Nc = lane&15.

// CDNA5 async global->LDS copy (ASYNCcnt path, ISA §10.7 / §15.18.3).
// Per-lane: 16 bytes from per-lane global addr to per-lane LDS byte offset.
static __device__ inline void async_copy_b128(void* lds, const void* gptr) {
  asm volatile("global_load_async_to_lds_b128 %0, %1, off"
               :: "v"((uint32_t)(size_t)lds), "v"(gptr)
               : "memory");
}
static __device__ inline void wait_asynccnt0() {
  asm volatile("s_wait_asynccnt 0" ::: "memory");
}

// ---------------------------------------------------------------------------
// Prepack kernels: one-time bf16 conversion / transpose. Weights live in the
// 192 MB L2 for the whole run, so this removes all f32->bf16 converts and all
// strided weight loads from the GEMM hot loops.
// ---------------------------------------------------------------------------
__global__ __launch_bounds__(256)
void pack_x_bf16(const float4* __restrict__ X, uint2* __restrict__ Xb)
{
  const size_t i = (size_t)blockIdx.x * 256 + threadIdx.x;  // 4 elems/thread
  const float4 f = X[i];
  union { uint2 u; __bf16 h[4]; } o;
  o.h[0] = (__bf16)f.x; o.h[1] = (__bf16)f.y;
  o.h[2] = (__bf16)f.z; o.h[3] = (__bf16)f.w;
  Xb[i] = o.u;
}

__global__ __launch_bounds__(256)
void pack_w_transpose_bf16(const float* __restrict__ W,  // [rows][cols]
                           __bf16* __restrict__ Wt,      // [cols][rows]
                           int rows, int cols)
{
  const int i = blockIdx.x * 256 + threadIdx.x;
  if (i >= rows * cols) return;
  const int r = i / cols, c = i % cols;
  Wt[(size_t)c * rows + r] = (__bf16)W[i];
}

// ---------------------------------------------------------------------------
// Kernel 1: qkv = x @ W_qkv + b ; spike = (qkv/2 - 1 >= 0)
// Wave computes a 16x64 strip (4 N-tiles): A-fragment loaded once per K-step.
// All fragment loads are contiguous 16 B from prepacked bf16 buffers.
// ---------------------------------------------------------------------------
__global__ __launch_bounds__(256)
void qkv_spike_gemm(const __bf16* __restrict__ Xb,   // [TOK, C] bf16
                    const __bf16* __restrict__ Wt,   // [C3, C] bf16 (transposed)
                    const float*  __restrict__ bias, // [C3]
                    __bf16* __restrict__ qs,         // [T*B*H, N, D]
                    __bf16* __restrict__ ks,
                    __bf16* __restrict__ vs)
{
  const int lane = threadIdx.x & 31;
  const int wave = threadIdx.x >> 5;
  const int NG   = C3 / 64;                 // 12 N-groups of 64 cols
  const int gt   = blockIdx.x * 8 + wave;
  const int tm   = gt / NG;
  const int ng   = gt % NG;
  const int half = lane >> 4;
  const int l15  = lane & 15;

  const __bf16* arow = Xb + (size_t)(tm * 16 + l15) * C;
  __builtin_prefetch(Wt + (size_t)(ng * 64 + l15) * C, 0, 0);

  v8f acc[4] = {{}, {}, {}, {}};
  for (int k0 = 0; k0 < C; k0 += 32) {
    Frag16 a;
    const u32x4* ab = (const u32x4*)(arow + k0 + half * 16);
    a.q[0] = ab[0];           // K = half*16 .. +7
    a.q[1] = ab[1];           // K = half*16 + 8 .. +7
#pragma unroll
    for (int t = 0; t < 4; ++t) {
      const u32x4* bb =
          (const u32x4*)(Wt + (size_t)(ng * 64 + t * 16 + l15) * C + k0 + half * 16);
      Frag16 b;
      b.q[0] = bb[0];
      b.q[1] = bb[1];
      acc[t] = wmma_bf16(a, b, acc[t]);
    }
  }

#pragma unroll
  for (int t = 0; t < 4; ++t) {
    const int   bcol  = ng * 64 + t * 16 + l15;
    const float bb    = bias[bcol];
    const int   which = bcol >> 8;           // 0=q 1=k 2=v
    const int   head  = (bcol & 255) >> 5;
    const int   dd    = bcol & 31;
    __bf16* dst = (which == 0) ? qs : (which == 1) ? ks : vs;
#pragma unroll
    for (int i = 0; i < 8; ++i) {
      const int   token = tm * 16 + i + 8 * half;
      const float y     = acc[t][i] + bb;
      const __bf16 sp   = (__bf16)((y >= 2.0f) ? 1.0f : 0.0f); // y/2 - 1 >= 0
      const int tb = token >> 10;
      const int n  = token & 1023;
      dst[((size_t)(tb * H + head) * N + n) * D + dd] = sp;
    }
  }
}

// ---------------------------------------------------------------------------
// Kernel 2: region sums + masked sim (8x8) + top-4 routing indices.
// ---------------------------------------------------------------------------
__global__ __launch_bounds__(64)
void routing_topk(const __bf16* __restrict__ qs,
                  const __bf16* __restrict__ ks,
                  int* __restrict__ idxbuf)    // [T*B*H, NW, TOPK]
{
  __shared__ float qreg[NW][D];
  __shared__ float kreg[NW][D];
  __shared__ float act[NW];
  __shared__ float sim[NW][NW];

  const int tbh = blockIdx.x;   // 0..255
  const int tid = threadIdx.x;  // 64 threads

  for (int w = tid; w < NW * D * 2; w += 64) {
    const int isK  = (w >= NW * D);
    const int r    = w - isK * NW * D;
    const int jw   = r / D;
    const int dcol = r % D;
    const __bf16* src = (isK ? ks : qs) + ((size_t)tbh * N + jw * WIN) * D + dcol;
    float s = 0.f;
    for (int rr = 0; rr < WIN; ++rr) s += (float)src[(size_t)rr * D];
    if (isK) kreg[jw][dcol] = s; else qreg[jw][dcol] = s;
  }
  __syncthreads();

  if (tid < NW) {
    float a = 0.f;
    for (int dcol = 0; dcol < D; ++dcol) a += kreg[tid][dcol];
    act[tid] = a;
  }
  __syncthreads();

  if (tid < NW * NW) {
    const int i = tid / NW, jw = tid % NW;
    float s = 0.f;
    for (int dcol = 0; dcol < D; ++dcol) s += qreg[i][dcol] * kreg[jw][dcol];
    s *= SCALE;
    if (!(act[jw] > 1e-5f)) s += -1e9f;
    sim[i][jw] = s;
  }
  __syncthreads();

  if (tid < NW) {
    unsigned taken = 0;
    for (int kk = 0; kk < TOPK; ++kk) {
      int best = 0; float bv = -3.4e38f;
      for (int jw = 0; jw < NW; ++jw)
        if (!((taken >> jw) & 1u) && sim[tid][jw] > bv) { bv = sim[tid][jw]; best = jw; }
      taken |= 1u << best;
      idxbuf[(tbh * NW + tid) * TOPK + kk] = best;
    }
  }
}

// ---------------------------------------------------------------------------
// Kernel 3: windowed sparse attention.
// Staging via global_load_async_to_lds_b128 (ASYNCcnt); 200 KB LDS panel.
// ---------------------------------------------------------------------------
__global__ __launch_bounds__(256)
void window_attention(const __bf16* __restrict__ qs,
                      const __bf16* __restrict__ ks,
                      const __bf16* __restrict__ vs,
                      const int*    __restrict__ idxbuf,
                      __bf16* __restrict__ ao)      // [T*B, N, C] bf16
{
  extern __shared__ char smem[];
  __bf16* sQ   = (__bf16*)smem;                  // 128*32
  __bf16* sK   = sQ + WIN * D;                   // 512*32
  __bf16* sV   = sK + KEYS * D;                  // 512*32
  __bf16* sS   = sV + KEYS * D;                  // 128*512 scores->probs
  float*  rsum = (float*)(sS + WIN * KEYS);      // 128 row sums

  const int blk = blockIdx.x;        // tbh*NW + j
  const int jw  = blk & (NW - 1);
  const int tbh = blk >> 3;
  const int hh  = tbh & (H - 1);
  const int tb  = tbh >> 3;

  // ---- async-stage Q window and gathered K/V windows into LDS ----
  {
    const __bf16* qsrc = qs + ((size_t)tbh * N + jw * WIN) * D;
    for (int ch = threadIdx.x; ch < WIN * D / 8; ch += 256)        // 512 x 16B
      async_copy_b128(sQ + ch * 8, qsrc + ch * 8);
    for (int kt = 0; kt < TOPK; ++kt) {
      const int wsrc = idxbuf[blk * TOPK + kt];
      const __bf16* kbase = ks + ((size_t)tbh * N + wsrc * WIN) * D;
      const __bf16* vbase = vs + ((size_t)tbh * N + wsrc * WIN) * D;
      __bf16* kd = sK + kt * WIN * D;
      __bf16* vd = sV + kt * WIN * D;
      for (int ch = threadIdx.x; ch < WIN * D / 8; ch += 256) {
        async_copy_b128(kd + ch * 8, kbase + ch * 8);
        async_copy_b128(vd + ch * 8, vbase + ch * 8);
      }
    }
    wait_asynccnt0();   // this thread's async transfers complete
  }
  __syncthreads();      // all threads' transfers visible

  const int lane = threadIdx.x & 31;
  const int wave = threadIdx.x >> 5;    // M-strip: rows [16*wave, 16*wave+16)
  const int half = lane >> 4;
  const int l15  = lane & 15;

  // ---- scores: S[strip, 0..511] = Q @ K^T * scale ----
  Frag16 aq;   // Q strip fragment, invariant over key tiles
  {
    const u32x4* base = (const u32x4*)(sQ + (wave * 16 + l15) * D + half * 16);
    aq.q[0] = base[0];
    aq.q[1] = base[1];
  }
  for (int nt = 0; nt < KEYS / 16; ++nt) {
    Frag16 bk;   // lane = key column; contiguous K run of 16 bf16 (32 B)
    const u32x4* kb = (const u32x4*)(sK + (nt * 16 + l15) * D + half * 16);
    bk.q[0] = kb[0];
    bk.q[1] = kb[1];
    v8f s = {};
    s = wmma_bf16(aq, bk, s);
#pragma unroll
    for (int i = 0; i < 8; ++i) {
      const int srow = wave * 16 + i + 8 * half;
      sS[(size_t)srow * KEYS + nt * 16 + l15] = (__bf16)(s[i] * SCALE);
    }
  }
  __syncthreads();

  // ---- softmax per score row (unnormalized probs; fold 1/sum into output) --
  if (threadIdx.x < WIN) {
    __bf16* rowp = sS + (size_t)threadIdx.x * KEYS;
    float mx = -3.4e38f;
    for (int c = 0; c < KEYS; ++c) mx = fmaxf(mx, (float)rowp[c]);
    float sum = 0.f;
    for (int c = 0; c < KEYS; ++c) {
      const float e = __expf((float)rowp[c] - mx);
      sum += e;
      rowp[c] = (__bf16)e;
    }
    rsum[threadIdx.x] = sum;
  }
  __syncthreads();

  // ---- out = P @ V  (K = 512, two 16-wide N tiles) ----
  v8f o0 = {}, o1 = {};
  for (int kstep = 0; kstep < KEYS / 32; ++kstep) {
    Frag16 ap;
    const u32x4* pb =
        (const u32x4*)(sS + (size_t)(wave * 16 + l15) * KEYS + kstep * 32 + half * 16);
    ap.q[0] = pb[0];
    ap.q[1] = pb[1];
    Frag16 bv0, bv1;
#pragma unroll
    for (int e = 0; e < 16; ++e) {
      const int krow = kstep * 32 + half * 16 + e;
      bv0.h[e] = sV[(size_t)krow * D + l15];
      bv1.h[e] = sV[(size_t)krow * D + 16 + l15];
    }
    o0 = wmma_bf16(ap, bv0, o0);
    o1 = wmma_bf16(ap, bv1, o1);
  }

#pragma unroll
  for (int i = 0; i < 8; ++i) {
    const int   row = wave * 16 + i + 8 * half;
    const float inv = 1.0f / rsum[row];
    const int   n   = jw * WIN + row;
    const size_t base = ((size_t)tb * N + n) * C + hh * D;
    ao[base + l15]      = (__bf16)(o0[i] * inv);
    ao[base + 16 + l15] = (__bf16)(o1[i] * inv);
  }
}

// ---------------------------------------------------------------------------
// Kernel 4: out = lif(attn_out @ W_proj + b_proj)  -> f32 {0,1}
// 16x64 strip per wave, prepacked transposed bf16 weights.
// ---------------------------------------------------------------------------
__global__ __launch_bounds__(256)
void proj_spike_gemm(const __bf16* __restrict__ A,    // [TOK, C] bf16
                     const __bf16* __restrict__ Wt,   // [C, C] bf16 (transposed)
                     const float*  __restrict__ bias, // [C]
                     float* __restrict__ out)         // [TOK, C]
{
  const int lane = threadIdx.x & 31;
  const int wave = threadIdx.x >> 5;
  const int NG   = C / 64;                  // 4 N-groups
  const int gt   = blockIdx.x * 8 + wave;
  const int tm   = gt / NG;
  const int ng   = gt % NG;
  const int half = lane >> 4;
  const int l15  = lane & 15;

  const __bf16* arow = A + (size_t)(tm * 16 + l15) * C;

  v8f acc[4] = {{}, {}, {}, {}};
  for (int k0 = 0; k0 < C; k0 += 32) {
    Frag16 a;
    const u32x4* ab = (const u32x4*)(arow + k0 + half * 16);
    a.q[0] = ab[0];
    a.q[1] = ab[1];
#pragma unroll
    for (int t = 0; t < 4; ++t) {
      const u32x4* bb =
          (const u32x4*)(Wt + (size_t)(ng * 64 + t * 16 + l15) * C + k0 + half * 16);
      Frag16 b;
      b.q[0] = bb[0];
      b.q[1] = bb[1];
      acc[t] = wmma_bf16(a, b, acc[t]);
    }
  }

#pragma unroll
  for (int t = 0; t < 4; ++t) {
    const int   bcol = ng * 64 + t * 16 + l15;
    const float bb   = bias[bcol];
#pragma unroll
    for (int i = 0; i < 8; ++i) {
      const int   token = tm * 16 + i + 8 * half;
      const float y     = acc[t][i] + bb;
      out[(size_t)token * C + bcol] = (y >= 2.0f) ? 1.0f : 0.0f;
    }
  }
}

// ---------------------------------------------------------------------------
// Host entry
// ---------------------------------------------------------------------------
extern "C" void kernel_launch(void* const* d_in, const int* in_sizes, int n_in,
                              void* d_out, int out_size, void* d_ws, size_t ws_size,
                              hipStream_t stream)
{
  (void)in_sizes; (void)n_in; (void)out_size; (void)ws_size;

  const float* x     = (const float*)d_in[0];
  const float* Wqkv  = (const float*)d_in[1];
  const float* bqkv  = (const float*)d_in[2];
  const float* Wproj = (const float*)d_in[3];
  const float* bproj = (const float*)d_in[4];
  float*       out   = (float*)d_out;

  // workspace carve-up
  char*  ws  = (char*)d_ws;
  size_t off = 0;
  __bf16* qs  = (__bf16*)(ws + off); off += (size_t)TOK * C * sizeof(__bf16);   // 16 MB
  __bf16* ks  = (__bf16*)(ws + off); off += (size_t)TOK * C * sizeof(__bf16);   // 16 MB
  __bf16* vs  = (__bf16*)(ws + off); off += (size_t)TOK * C * sizeof(__bf16);   // 16 MB
  int*    ix  = (int*)   (ws + off); off += (size_t)T * B * H * NW * TOPK * sizeof(int);
  off = (off + 255) & ~(size_t)255;
  __bf16* ao  = (__bf16*)(ws + off); off += (size_t)TOK * C * sizeof(__bf16);   // 16 MB
  __bf16* Xb  = (__bf16*)(ws + off); off += (size_t)TOK * C * sizeof(__bf16);   // 16 MB
  __bf16* Wqt = (__bf16*)(ws + off); off += (size_t)C3 * C * sizeof(__bf16);    // 384 KB
  __bf16* Wpt = (__bf16*)(ws + off); off += (size_t)C * C * sizeof(__bf16);     // 128 KB

  // 0) prepack: x -> bf16; weights -> bf16 transposed [n][k]
  pack_x_bf16<<<TOK * C / 4 / 256, 256, 0, stream>>>((const float4*)x, (uint2*)Xb);
  pack_w_transpose_bf16<<<(C * C3 + 255) / 256, 256, 0, stream>>>(Wqkv, Wqt, C, C3);
  pack_w_transpose_bf16<<<(C * C + 255) / 256, 256, 0, stream>>>(Wproj, Wpt, C, C);

  // 1) QKV projection + LIF spikes (bf16 WMMA, 16x64 strip per wave)
  qkv_spike_gemm<<<(TOK / 16) * (C3 / 64) / 8, 256, 0, stream>>>(
      Xb, Wqt, bqkv, qs, ks, vs);

  // 2) routing: region sums + masked sim + top-4
  routing_topk<<<T * B * H, 64, 0, stream>>>(qs, ks, ix);

  // 3) windowed sparse attention (async-to-LDS staging + bf16 WMMA)
  const size_t smem =
      ((size_t)WIN * D + 2 * (size_t)KEYS * D + (size_t)WIN * KEYS) * sizeof(__bf16)
      + WIN * sizeof(float);   // ~200.5 KB < 320 KB/WGP
  window_attention<<<T * B * H * NW, 256, smem, stream>>>(qs, ks, vs, ix, ao);

  // 4) projection + LIF (bf16 WMMA), final f32 {0,1} output
  proj_spike_gemm<<<(TOK / 16) * (C / 64) / 8, 256, 0, stream>>>(
      ao, Wpt, bproj, out);
}
